// HNEPY_52999896433115
// MI455X (gfx1250) — compile-verified, matrix operand order checked by reading, not run on hardware
//
#include <hip/hip_runtime.h>

// ---------------- problem dims ----------------
#define NN1 4000
#define NN2 6000
#define NN3 4000
#define NTOT 14000
#define DD1 1024
#define DD2 512
#define DD3 256
#define RR1 16
#define RR2 32
#define EE 100000
#define KPAD 14016            // 14000 padded up to multiple of 32
#define KCHUNKS 438           // ceil(14000/32); last chunk has 16 valid K
#define KSPLIT 16
#define CHUNKS_PER_SPLIT 28   // 15*28 + 18 = 438
#define MTILES 875            // 14000/16
#define ETILES 875            // 250 + 375 + 250 embedding tiles

typedef __attribute__((ext_vector_type(16))) __bf16 v16bf;
typedef __attribute__((ext_vector_type(8)))  float  v8f;

union V16BF { v16bf v; __bf16 b[16]; unsigned short u[16]; unsigned int w[8]; };

__device__ __forceinline__ unsigned short f2bf(float f) {
  union { __bf16 h; unsigned short u; } c; c.h = (__bf16)f; return c.u;
}

// Pack A-fragment (16-bit A 16x32 layout): elements 0..7 = K+0..7, 8..15 = K+16..23
__device__ __forceinline__ void packA(V16BF& a, const float* p0, const float* p1,
                                      bool v0, bool v1) {
  if (v0) {
    float4 x = *(const float4*)p0; float4 y = *(const float4*)(p0 + 4);
    a.b[0]=(__bf16)x.x; a.b[1]=(__bf16)x.y; a.b[2]=(__bf16)x.z; a.b[3]=(__bf16)x.w;
    a.b[4]=(__bf16)y.x; a.b[5]=(__bf16)y.y; a.b[6]=(__bf16)y.z; a.b[7]=(__bf16)y.w;
  } else { a.w[0]=0u; a.w[1]=0u; a.w[2]=0u; a.w[3]=0u; }
  if (v1) {
    float4 x = *(const float4*)p1; float4 y = *(const float4*)(p1 + 4);
    a.b[8]=(__bf16)x.x;  a.b[9]=(__bf16)x.y;  a.b[10]=(__bf16)x.z; a.b[11]=(__bf16)x.w;
    a.b[12]=(__bf16)y.x; a.b[13]=(__bf16)y.y; a.b[14]=(__bf16)y.z; a.b[15]=(__bf16)y.w;
  } else { a.w[4]=0u; a.w[5]=0u; a.w[6]=0u; a.w[7]=0u; }
}

// Pack B-fragment from pre-transposed bf16 buffer (contiguous 16B per half)
__device__ __forceinline__ void packB(V16BF& b, const unsigned short* p0,
                                      const unsigned short* p1) {
  uint4 x = *(const uint4*)p0; uint4 y = *(const uint4*)p1;
  b.w[0]=x.x; b.w[1]=x.y; b.w[2]=x.z; b.w[3]=x.w;
  b.w[4]=y.x; b.w[5]=y.y; b.w[6]=y.z; b.w[7]=y.w;
}

// ---------------- init: zero accumulators / scalars / output ----------------
__global__ void k_init(float* Hpre, float* sumSe0, float* out) {
  int gid = blockIdx.x * blockDim.x + threadIdx.x;
  if (gid < NTOT * RR2) Hpre[gid] = 0.0f;
  if (gid == 0) { sumSe0[0] = 0.0f; out[0] = 0.0f; }
}

// ---------------- transpose encoder weights to bf16 (Wt[c][k]) ----------------
__global__ void k_wt(const float* W1, const float* W2, const float* W3,
                     unsigned short* Wt1, unsigned short* Wt2, unsigned short* Wt3) {
  int tid = blockIdx.x * blockDim.x + threadIdx.x;
  if (tid < 16 * DD1) {
    int c = tid / DD1, k = tid % DD1;
    Wt1[tid] = f2bf(W1[k * 16 + c]);
  } else if (tid < 16 * (DD1 + DD2)) {
    int t = tid - 16 * DD1; int c = t / DD2, k = t % DD2;
    Wt2[t] = f2bf(W2[k * 16 + c]);
  } else if (tid < 16 * (DD1 + DD2 + DD3)) {
    int t = tid - 16 * (DD1 + DD2); int c = t / DD3, k = t % DD3;
    Wt3[t] = f2bf(W3[k * 16 + c]);
  }
}

// ---------------- stage 1: X = tanh(fea @ W_e + b) via bf16 WMMA ----------------
__global__ void k_embed(const float* f1, const float* f2, const float* f3,
                        const unsigned short* Wt1, const unsigned short* Wt2,
                        const unsigned short* Wt3,
                        const float* b1, const float* b2, const float* b3,
                        float* X) {
  int wid = threadIdx.x >> 5, lane = threadIdx.x & 31;
  int t = blockIdx.x * 4 + wid;
  if (t >= ETILES) return;                       // uniform per wave
  const float* fea; const unsigned short* Wt; const float* bias;
  int D, rowBase, lt;
  if (t < 250)      { lt = t;       fea = f1; Wt = Wt1; bias = b1; D = DD1; rowBase = 0; }
  else if (t < 625) { lt = t - 250; fea = f2; Wt = Wt2; bias = b2; D = DD2; rowBase = NN1; }
  else              { lt = t - 625; fea = f3; Wt = Wt3; bias = b3; D = DD3; rowBase = NN1 + NN2; }

  int mloc = lt * 16 + (lane & 15);
  int koff = (lane >> 4) * 8;
  int col  = lane & 15;
  const float* arow = fea + (size_t)mloc * D;
  const unsigned short* brow = Wt + (size_t)col * D;

  v8f acc = {};
  for (int kb = 0; kb < D; kb += 32) {
    V16BF a, b;
    packA(a, arow + kb + koff, arow + kb + koff + 16, true, true);
    packB(b, brow + kb + koff, brow + kb + koff + 16);
    acc = __builtin_amdgcn_wmma_f32_16x16x32_bf16(false, a.v, false, b.v,
                                                  (short)0, acc, false, false);
  }
  float bb = bias[col];
  int rtop = (lane >> 4) * 8;
  for (int i = 0; i < 8; ++i) {
    int row = rowBase + lt * 16 + rtop + i;
    X[(size_t)row * 16 + col] = tanhf(acc[i] + bb);
  }
}

// ---------------- stage 2: Pt = (X @ Wg1)^T in bf16, K zero-padded ----------------
__global__ void k_p(const float* X, const float* Wg1, unsigned short* Pt) {
  int tid = blockIdx.x * blockDim.x + threadIdx.x;
  if (tid >= KPAD * 32) return;
  int r = tid >> 5, j = tid & 31;
  unsigned short o = 0;
  if (r < NTOT) {
    float s = 0.0f;
    for (int k = 0; k < 16; ++k) s += X[(size_t)r * 16 + k] * Wg1[k * 32 + j];
    o = f2bf(s);
  }
  Pt[(size_t)j * KPAD + r] = o;
}

// ---------------- stage 3: Hpre += A @ P  (the 784 MB stream; bf16 WMMA) ----------
__global__ void k_bigmm(const float* A, const unsigned short* Pt, float* Hpre) {
  int wid = threadIdx.x >> 5, lane = threadIdx.x & 31;
  int gw = blockIdx.x * 4 + wid;          // 14000 waves total, exact grid
  int mTile = gw >> 4, ks = gw & 15;
  int cBeg = ks * CHUNKS_PER_SPLIT;
  int cEnd = cBeg + CHUNKS_PER_SPLIT; if (cEnd > KCHUNKS) cEnd = KCHUNKS;

  int r    = mTile * 16 + (lane & 15);
  int koff = (lane >> 4) * 8;
  int col  = lane & 15;
  const float* arow = A + (size_t)r * NTOT;
  const unsigned short* b0row = Pt + (size_t)col * KPAD;
  const unsigned short* b1row = Pt + (size_t)(col + 16) * KPAD;

  v8f c0 = {}, c1 = {};
#pragma unroll 2
  for (int c = cBeg; c < cEnd; ++c) {
    int k0 = c * 32 + koff, k1 = k0 + 16;
    if (c + 1 < cEnd)
      __builtin_prefetch(arow + (c + 1) * 32 + koff, 0, 0);   // global_prefetch_b8
    V16BF a, b0, b1;
    packA(a, arow + k0, arow + k1, k0 < NTOT, k1 < NTOT);     // zero-fill K tail
    packB(b0, b0row + k0, b0row + k1);                        // Pt zero-padded
    packB(b1, b1row + k0, b1row + k1);
    c0 = __builtin_amdgcn_wmma_f32_16x16x32_bf16(false, a.v, false, b0.v,
                                                 (short)0, c0, false, false);
    c1 = __builtin_amdgcn_wmma_f32_16x16x32_bf16(false, a.v, false, b1.v,
                                                 (short)0, c1, false, false);
  }
  int rtop = (lane >> 4) * 8;
  for (int i = 0; i < 8; ++i) {
    int row = mTile * 16 + rtop + i;
    unsafeAtomicAdd(&Hpre[(size_t)row * 32 + col],      c0[i]);  // global_atomic_add_f32
    unsafeAtomicAdd(&Hpre[(size_t)row * 32 + col + 16], c1[i]);
  }
}

// ---------------- stage 4: emb = tanh(Hpre + bg1) @ Wg2 + bg2 ----------------
__global__ void k_emb(const float* Hpre, const float* bg1, const float* Wg2,
                      const float* bg2, float* emb) {
  int row = blockIdx.x * blockDim.x + threadIdx.x;
  if (row >= NTOT) return;
  float h[32];
  for (int j = 0; j < 32; ++j) h[j] = tanhf(Hpre[(size_t)row * 32 + j] + bg1[j]);
  for (int c = 0; c < 16; ++c) {
    float s = bg2[c];
    for (int j = 0; j < 32; ++j) s += h[j] * Wg2[j * 16 + c];
    emb[(size_t)row * 16 + c] = s;
  }
}

// ---------------- stage 5: edge scoring (constants staged in LDS) ----------------
struct ScoreSmem {
  float B1[256];
  float B2m[256];
  float W_B2[48];
  float b_B2[3];
  float b_lin[3];
  float W_sim[3];
  float b_sim;
  float red[256];
};

__device__ __forceinline__ void stage_consts(ScoreSmem& s,
    const float* B1, const float* B2m, const float* W_B2, const float* b_B2,
    const float* b_lin, const float* W_sim, const float* b_sim) {
  int t = threadIdx.x;
  s.B1[t]  = B1[t];
  s.B2m[t] = B2m[t];
  if (t < 48) s.W_B2[t] = W_B2[t];
  if (t < 3)  { s.b_B2[t] = b_B2[t]; s.b_lin[t] = b_lin[t]; s.W_sim[t] = W_sim[t]; }
  if (t == 0) s.b_sim = b_sim[0];
  __syncthreads();
}

__device__ __forceinline__ float edge_score(const int* e, const float* emb,
                                            const ScoreSmem& s) {
  int di = e[1], ii = e[3], ai = e[5];
  const float* dr = emb + (size_t)di * 16;
  const float* in = emb + (size_t)(NN1 + ii) * 16;
  const float* ad = emb + (size_t)(NN2 + ai) * 16;   // reference slices d3_eb = emb[N2:N2+N3]
  float b1 = 0.0f, b2 = 0.0f;
  for (int r = 0; r < 16; ++r) {
    float d = dr[r];
    for (int c = 0; c < 16; ++c) {
      b1 += d * s.B1[r * 16 + c]  * in[c];
      b2 += d * s.B2m[r * 16 + c] * ad[c];
    }
  }
  float sim = s.b_sim;
  for (int k = 0; k < 3; ++k) {
    float t = 0.0f;
    for (int r = 0; r < 16; ++r) t += (dr[r] + in[r] + ad[r]) * s.W_B2[r * 3 + k];
    t = t * (1.0f / 3.0f) + s.b_B2[k];
    float bm = (k == 0) ? b1 : ((k == 1) ? b2 : 0.0f);
    sim += tanhf(bm + t + s.b_lin[k]) * s.W_sim[k];
  }
  return sim;
}

__global__ void k_neg(const int* edges, const float* emb,
    const float* B1, const float* B2m, const float* W_B2, const float* b_B2,
    const float* b_lin, const float* W_sim, const float* b_sim, float* sumSe0) {
  __shared__ ScoreSmem s;
  stage_consts(s, B1, B2m, W_B2, b_B2, b_lin, W_sim, b_sim);
  int e = blockIdx.x * 256 + threadIdx.x;
  float v = (e < EE) ? edge_score(edges + (size_t)e * 6, emb, s) : 0.0f;
  s.red[threadIdx.x] = v; __syncthreads();
  for (int st = 128; st > 0; st >>= 1) {
    if (threadIdx.x < st) s.red[threadIdx.x] += s.red[threadIdx.x + st];
    __syncthreads();
  }
  if (threadIdx.x == 0) unsafeAtomicAdd(sumSe0, s.red[0]);
}

__global__ void k_pos(const int* edges, const float* emb,
    const float* B1, const float* B2m, const float* W_B2, const float* b_B2,
    const float* b_lin, const float* W_sim, const float* b_sim,
    const float* sumSe0, float* out) {
  __shared__ ScoreSmem s;
  stage_consts(s, B1, B2m, W_B2, b_B2, b_lin, W_sim, b_sim);
  int e = blockIdx.x * 256 + threadIdx.x;
  float mS0 = sumSe0[0] * (1.0f / (float)EE);
  float v = 0.0f;
  if (e < EE) {
    float se = edge_score(edges + (size_t)e * 6, emb, s);
    float x = mS0 - se;
    v = (x > 20.0f) ? x : log1pf(expf(x));       // stable softplus
  }
  s.red[threadIdx.x] = v; __syncthreads();
  for (int st = 128; st > 0; st >>= 1) {
    if (threadIdx.x < st) s.red[threadIdx.x] += s.red[threadIdx.x + st];
    __syncthreads();
  }
  if (threadIdx.x == 0) unsafeAtomicAdd(out, s.red[0] * (1.0f / (float)EE));
}

// ---------------- host launcher ----------------
extern "C" void kernel_launch(void* const* d_in, const int* in_sizes, int n_in,
                              void* d_out, int out_size, void* d_ws, size_t ws_size,
                              hipStream_t stream) {
  const int*   pos_e = (const int*)d_in[0];
  const int*   neg_e = (const int*)d_in[1];
  const float* f1    = (const float*)d_in[2];
  const float* f2    = (const float*)d_in[3];
  const float* f3    = (const float*)d_in[4];
  const float* A     = (const float*)d_in[5];
  const float* W_e1  = (const float*)d_in[6];
  const float* b_e1  = (const float*)d_in[7];
  const float* W_e2  = (const float*)d_in[8];
  const float* b_e2  = (const float*)d_in[9];
  const float* W_e3  = (const float*)d_in[10];
  const float* b_e3  = (const float*)d_in[11];
  const float* Wg1   = (const float*)d_in[12];
  const float* bg1   = (const float*)d_in[13];
  const float* Wg2   = (const float*)d_in[14];
  const float* bg2   = (const float*)d_in[15];
  const float* B1    = (const float*)d_in[16];
  const float* B2m   = (const float*)d_in[17];
  const float* W_B2  = (const float*)d_in[18];
  const float* b_B2  = (const float*)d_in[19];
  const float* b_lin = (const float*)d_in[20];
  const float* W_sim = (const float*)d_in[21];
  const float* b_sim = (const float*)d_in[22];
  float* out = (float*)d_out;

  // workspace layout (≈4.6 MB)
  unsigned char* ws = (unsigned char*)d_ws;
  unsigned short* Wt1 = (unsigned short*)(ws + 0);            // 16*1024 bf16
  unsigned short* Wt2 = Wt1 + 16 * DD1;                       // 16*512
  unsigned short* Wt3 = Wt2 + 16 * DD2;                       // 16*256  (ends 57344)
  float*          X    = (float*)(ws + 57344);                // 14000*16 f32
  unsigned short* Pt   = (unsigned short*)(ws + 953344);      // 32*14016 bf16
  float*          Hpre = (float*)(ws + 1850368);              // 14000*32 f32
  float*          emb  = (float*)(ws + 3642368);              // 14000*16 f32
  float*          sumSe0 = (float*)(ws + 4538368);            // 1 f32

  k_init <<<(NTOT * 32 + 255) / 256, 256, 0, stream>>>(Hpre, sumSe0, out);
  k_wt   <<<(16 * (DD1 + DD2 + DD3) + 255) / 256, 256, 0, stream>>>(
           W_e1, W_e2, W_e3, Wt1, Wt2, Wt3);
  k_embed<<<(ETILES + 3) / 4, 128, 0, stream>>>(
           f1, f2, f3, Wt1, Wt2, Wt3, b_e1, b_e2, b_e3, X);
  k_p    <<<(KPAD * 32 + 255) / 256, 256, 0, stream>>>(X, Wg1, Pt);
  k_bigmm<<<(MTILES * KSPLIT) / 4, 128, 0, stream>>>(A, Pt, Hpre);
  k_emb  <<<(NTOT + 255) / 256, 256, 0, stream>>>(Hpre, bg1, Wg2, bg2, emb);
  k_neg  <<<(EE + 255) / 256, 256, 0, stream>>>(
           neg_e, emb, B1, B2m, W_B2, b_B2, b_lin, W_sim, b_sim, sumSe0);
  k_pos  <<<(EE + 255) / 256, 256, 0, stream>>>(
           pos_e, emb, B1, B2m, W_B2, b_B2, b_lin, W_sim, b_sim, sumSe0, out);
}
